// KMeans_34746285425110
// MI455X (gfx1250) — compile-verified, matrix operand order checked by reading, not run on hardware
//
#include <hip/hip_runtime.h>
#include <hip/hip_bf16.h>

typedef __attribute__((ext_vector_type(16))) __bf16 v16bf;
typedef __attribute__((ext_vector_type(8)))  __bf16 v8bf;
typedef __attribute__((ext_vector_type(8)))  float  v8f;
typedef __attribute__((ext_vector_type(4)))  float  v4f;

#define D_DIM      256
#define K_CL       512
#define LDS_STRIDE 264   // 132 dwords/row: lanes 0..15 hit bank groups 4c..4c+3 -> conflict-free b128 reads

// ---------------- zero workspace (sums + counts) ----------------
__global__ void zero_ws_kernel(float* __restrict__ sums, float* __restrict__ counts) {
    int i = blockIdx.x * blockDim.x + threadIdx.x;
    if (i < K_CL * D_DIM) sums[i] = 0.0f;
    if (i < K_CL)         counts[i] = 0.0f;
}

// ---------------- c2[k] = ||centers_k||^2 ----------------
__global__ void c2_kernel(const float* __restrict__ centers, float* __restrict__ c2) {
    int k = blockIdx.x * blockDim.x + threadIdx.x;
    if (k >= K_CL) return;
    const float* p = centers + (long)k * D_DIM;
    float s = 0.0f;
    for (int d = 0; d < D_DIM; ++d) { float v = p[d]; s += v * v; }
    c2[k] = s;
}

// Load 16 f32 (8 at p, 8 at p+16, non-temporal) -> bf16 WMMA operand chunk (ISA 16-bit A layout)
__device__ inline v16bf cvt_chunk_global(const float* p) {
    v4f f0 = __builtin_nontemporal_load((const v4f*)p);
    v4f f1 = __builtin_nontemporal_load((const v4f*)p + 1);
    v4f f2 = __builtin_nontemporal_load((const v4f*)(p + 16));
    v4f f3 = __builtin_nontemporal_load((const v4f*)(p + 16) + 1);
    v16bf r;
    r[0]  = (__bf16)f0[0]; r[1]  = (__bf16)f0[1]; r[2]  = (__bf16)f0[2]; r[3]  = (__bf16)f0[3];
    r[4]  = (__bf16)f1[0]; r[5]  = (__bf16)f1[1]; r[6]  = (__bf16)f1[2]; r[7]  = (__bf16)f1[3];
    r[8]  = (__bf16)f2[0]; r[9]  = (__bf16)f2[1]; r[10] = (__bf16)f2[2]; r[11] = (__bf16)f2[3];
    r[12] = (__bf16)f3[0]; r[13] = (__bf16)f3[1]; r[14] = (__bf16)f3[2]; r[15] = (__bf16)f3[3];
    return r;
}

union BV { v16bf v; v8bf h[2]; };

// ---------------- assignment: argmin_k (c2[k] - 2 x.c_k) via bf16 WMMA ----------------
// Each wave handles M=32 rows (two 16-row A tiles sharing every B load: 2x WMMA per KB of LDS).
__global__ __launch_bounds__(256) void assign_kernel(
        const float* __restrict__ x,
        const float* __restrict__ centers,
        const float* __restrict__ c2,
        int* __restrict__ cluster_out,
        int nTiles32) {
    extern __shared__ unsigned char smem_raw[];
    __bf16* sc  = (__bf16*)smem_raw;                                   // [K_CL][LDS_STRIDE] bf16
    float*  sc2 = (float*)(smem_raw + (size_t)K_CL * LDS_STRIDE * 2);  // [K_CL]

    const int tid = threadIdx.x;
    // Stage all centers into LDS as bf16: v4f x2 reads -> one b128 LDS store (8 bf16)
    for (int i = tid * 8; i < K_CL * D_DIM; i += 256 * 8) {
        int k = i >> 8;
        int d = i & 255;
        v4f f0 = ((const v4f*)(centers + i))[0];
        v4f f1 = ((const v4f*)(centers + i))[1];
        v8bf pv;
        pv[0] = (__bf16)f0[0]; pv[1] = (__bf16)f0[1]; pv[2] = (__bf16)f0[2]; pv[3] = (__bf16)f0[3];
        pv[4] = (__bf16)f1[0]; pv[5] = (__bf16)f1[1]; pv[6] = (__bf16)f1[2]; pv[7] = (__bf16)f1[3];
        *(v8bf*)(sc + k * LDS_STRIDE + d) = pv;
    }
    for (int i = tid; i < K_CL; i += 256) sc2[i] = c2[i];
    __syncthreads();

    const int lane = tid & 31;
    const int wave = tid >> 5;
    const int half = lane >> 4;     // 0: lanes 0-15, 1: lanes 16-31
    const int lrow = lane & 15;     // M (A) / N (B) index within a 16-wide tile
    const int off  = half * 8;      // K-offset split per ISA 16-bit operand layout

    for (int tile = blockIdx.x * 8 + wave; tile < nTiles32; tile += gridDim.x * 8) {
        const float* xp0 = x + (long)(tile * 32 + lrow) * D_DIM;        // rows tile*32   .. +15
        const float* xp1 = xp0 + 16L * D_DIM;                           // rows tile*32+16.. +31

        v16bf a0[8], a1[8];
        #pragma unroll
        for (int c = 0; c < 8; ++c) {
            a0[c] = cvt_chunk_global(xp0 + c * 32 + off);
            a1[c] = cvt_chunk_global(xp1 + c * 32 + off);
        }

        float bestv0[8], bestv1[8];
        int   besti0[8], besti1[8];
        #pragma unroll
        for (int j = 0; j < 8; ++j) {
            bestv0[j] = 3.4e38f; besti0[j] = 0;
            bestv1[j] = 3.4e38f; besti1[j] = 0;
        }

        for (int t = 0; t < K_CL / 16; ++t) {
            const int col = t * 16 + lrow;
            const __bf16* cp = sc + col * LDS_STRIDE + off;

            v8f acc0 = {0.f, 0.f, 0.f, 0.f, 0.f, 0.f, 0.f, 0.f};
            v8f acc1 = {0.f, 0.f, 0.f, 0.f, 0.f, 0.f, 0.f, 0.f};

            // Software-pipelined B chunks: issue chunk c+1 loads before consuming chunk c.
            BV bcur;
            bcur.h[0] = *(const v8bf*)(cp);
            bcur.h[1] = *(const v8bf*)(cp + 16);
            #pragma unroll
            for (int c = 0; c < 8; ++c) {
                BV bnxt;
                if (c < 7) {
                    bnxt.h[0] = *(const v8bf*)(cp + (c + 1) * 32);
                    bnxt.h[1] = *(const v8bf*)(cp + (c + 1) * 32 + 16);
                }
                acc0 = __builtin_amdgcn_wmma_f32_16x16x32_bf16(
                        false, a0[c], false, bcur.v, (short)0, acc0, false, false);
                acc1 = __builtin_amdgcn_wmma_f32_16x16x32_bf16(
                        false, a1[c], false, bcur.v, (short)0, acc1, false, false);
                if (c < 7) bcur = bnxt;
            }

            const float cc = sc2[col];
            #pragma unroll
            for (int j = 0; j < 8; ++j) {
                float s0 = cc - 2.0f * acc0[j];   // x^2 is row-constant: irrelevant for argmin
                float s1 = cc - 2.0f * acc1[j];
                if (s0 < bestv0[j]) { bestv0[j] = s0; besti0[j] = col; }
                if (s1 < bestv1[j]) { bestv1[j] = s1; besti1[j] = col; }
            }
        }

        // Cross-lane argmin within each 16-lane half (cols live across lanes)
        #pragma unroll
        for (int j = 0; j < 8; ++j) {
            float v0 = bestv0[j]; int i0 = besti0[j];
            float v1 = bestv1[j]; int i1 = besti1[j];
            #pragma unroll
            for (int m = 8; m >= 1; m >>= 1) {
                float ov0 = __shfl_xor(v0, m, 32); int oi0 = __shfl_xor(i0, m, 32);
                float ov1 = __shfl_xor(v1, m, 32); int oi1 = __shfl_xor(i1, m, 32);
                if (ov0 < v0 || (ov0 == v0 && oi0 < i0)) { v0 = ov0; i0 = oi0; }
                if (ov1 < v1 || (ov1 == v1 && oi1 < i1)) { v1 = ov1; i1 = oi1; }
            }
            if (lrow == 0) {
                cluster_out[tile * 32 + j + half * 8]      = i0;  // rows j / j+8 of tile A0
                cluster_out[tile * 32 + 16 + j + half * 8] = i1;  // rows j / j+8 of tile A1
            }
        }
    }
}

// ---------------- per-cluster accumulation (one wave per point) ----------------
__global__ void accum_kernel(const float* __restrict__ x, const int* __restrict__ cl,
                             float* __restrict__ sums, float* __restrict__ counts, int n) {
    const int p    = (blockIdx.x * blockDim.x + threadIdx.x) >> 5;
    const int lane = threadIdx.x & 31;
    if (p >= n) return;
    const int c = cl[p];
    if (lane == 0) atomicAdd(&counts[c], 1.0f);
    const float* xp = x + (long)p * D_DIM;
    float* sp = sums + (long)c * D_DIM;
    #pragma unroll
    for (int i = 0; i < D_DIM / 32; ++i) {
        int d = lane + i * 32;
        float v = __builtin_nontemporal_load(xp + d);
        atomicAdd(&sp[d], v);
    }
}

// ---------------- EMA center update ----------------
__global__ void finalize_kernel(const float* __restrict__ centers, const float* __restrict__ sums,
                                const float* __restrict__ counts, float* __restrict__ out) {
    int i = blockIdx.x * blockDim.x + threadIdx.x;
    if (i >= K_CL * D_DIM) return;
    int k = i >> 8;
    float cnt  = counts[k];
    float c    = centers[i];
    float mean = sums[i] / fmaxf(cnt, 1.0f);
    float upd  = 0.8f * mean + 0.2f * c;
    out[i] = (cnt > 0.0f) ? upd : c;
}

extern "C" void kernel_launch(void* const* d_in, const int* in_sizes, int n_in,
                              void* d_out, int out_size, void* d_ws, size_t ws_size,
                              hipStream_t stream) {
    const float* x       = (const float*)d_in[0];
    const float* centers = (const float*)d_in[1];
    const int n = in_sizes[0] / D_DIM;           // 262144 points

    int*   cluster_out = (int*)d_out;            // output 0: x_cluster [N] (int32 bits)
    float* centers_out = (float*)d_out + n;      // output 1: new_centers [K,D] f32

    float* sums   = (float*)d_ws;                // K*D
    float* counts = sums + K_CL * D_DIM;         // K
    float* c2     = counts + K_CL;               // K

    zero_ws_kernel<<<(K_CL * D_DIM + 255) / 256, 256, 0, stream>>>(sums, counts);
    c2_kernel<<<(K_CL + 255) / 256, 256, 0, stream>>>(centers, c2);

    const int nTiles32 = n / 32;                 // 8192 32-row tiles
    const size_t shmem = (size_t)K_CL * LDS_STRIDE * sizeof(__bf16) + (size_t)K_CL * sizeof(float);
    (void)hipFuncSetAttribute((const void*)assign_kernel,
                              hipFuncAttributeMaxDynamicSharedMemorySize, (int)shmem);
    assign_kernel<<<256, 256, shmem, stream>>>(x, centers, c2, cluster_out, nTiles32);

    accum_kernel<<<n / 8, 256, 0, stream>>>(x, cluster_out, sums, counts, n);
    finalize_kernel<<<(K_CL * D_DIM + 255) / 256, 256, 0, stream>>>(centers, sums, counts, centers_out);
}
